// PolicyNet_13417477833138
// MI455X (gfx1250) — matrix operand order repaired and checked by reading, not compiled, not run on hardware
//
#include <hip/hip_runtime.h>
#include <hip/hip_bf16.h>

typedef __attribute__((ext_vector_type(16))) _Float16 v16h;
typedef __attribute__((ext_vector_type(8)))  _Float16 v8h;
typedef __attribute__((ext_vector_type(8)))  float    v8f;

// Global-address-space pointer types: force global_load_b128 (LOADcnt only)
// instead of flat_load_b128 (LOADcnt+DScnt) for the laundered weight stream.
typedef const __attribute__((address_space(1))) _Float16* gf16p;
typedef const __attribute__((address_space(1))) v16h*     gv16p;

#define B_      512
#define S_      512
#define D_      128
#define H_      256
#define IN_DIM_ 641
#define G3_     768   // 3*H

// ---------------------------------------------------------------------------
// Prep: convert weights to f16 once. wx = w_ih[:, 0:128] (768x128 row-major),
// whh = w_hh (768x256 row-major). The one-hot part of w_ih stays f32 and is
// gathered per (row, tag) at runtime.
// ---------------------------------------------------------------------------
__global__ void prep_weights_kernel(const float* __restrict__ w_ih,
                                    const float* __restrict__ w_hh,
                                    _Float16* __restrict__ wx,
                                    _Float16* __restrict__ whh) {
  int idx = blockIdx.x * blockDim.x + threadIdx.x;
  const int NX = G3_ * D_;   // 98304
  const int NH = G3_ * H_;   // 196608
  if (idx < NX) {
    int n = idx / D_, k = idx % D_;
    wx[idx] = (_Float16)w_ih[n * IN_DIM_ + k];
  } else if (idx < NX + NH) {
    int i2 = idx - NX;
    whh[i2] = (_Float16)w_hh[i2];
  }
}

// Build a 16x32 f16 A-fragment from a row-major LDS tile, in contiguous k
// order, matching the B loads: lanes 0-15 / 16-31 hold M=0..15 with K
// {0..15} / {16..31}. Two ds_load_b128 per fragment.
static __device__ __forceinline__ v16h load_a_frag(const _Float16* base, int ld,
                                                   int k0, int lane) {
  int m  = lane & 15;
  int kb = k0 + ((lane < 16) ? 0 : 16);
  union { v16h v; v8h h[2]; } u;
  u.h[0] = *(const v8h*)(base + m * ld + kb);
  u.h[1] = *(const v8h*)(base + m * ld + kb + 8);
  return u.v;
}

// ---------------------------------------------------------------------------
// Persistent recurrent kernel: 32 blocks x 512 threads (16 waves).
// Block b owns batch rows [16b, 16b+16) for all 512 timesteps.
// Wave w owns gate-column slice [w*16, w*16+16) of r, z and n (1 N-tile per
// gate => 3 gi + 3 gh accumulators), so the whole GRU gate update happens in
// registers with no gi/gh staging in LDS. Weights stream from L2 each step
// (pointer laundered per iteration so LICM cannot hoist+spill them).
// ---------------------------------------------------------------------------
__global__ __launch_bounds__(512, 1)
void policy_gru_kernel(const float* __restrict__ inputs,   // (B,S,D)
                       const float* __restrict__ w_ih,     // (768,641) f32
                       const float* __restrict__ b_ih,     // (768)
                       const float* __restrict__ b_hh,     // (768)
                       const float* __restrict__ w_cf,     // (2,256)
                       const float* __restrict__ b_cf,     // (2)
                       const _Float16* __restrict__ wx,    // (768,128) f16
                       const _Float16* __restrict__ whh,   // (768,256) f16
                       float* __restrict__ out) {          // 3*B*S f32
  __shared__ _Float16 sh_x[16 * D_];      // x_t tile, f16 row-major
  __shared__ _Float16 sh_h[16 * H_];      // h, f16 (GEMM operand)
  __shared__ float    sh_hf[16 * H_];     // h, f32 master copy
  __shared__ float    sh_logits[16 * 2];
  __shared__ int      sh_tag[16];

  const int tid  = threadIdx.x;
  const int lane = tid & 31;
  const int w    = tid >> 5;          // wave id 0..15
  const int b0   = blockIdx.x * 16;   // first batch row of this block

  for (int i = tid; i < 16 * H_; i += 512) { sh_hf[i] = 0.0f; sh_h[i] = (_Float16)0.0f; }
  if (tid < 16) sh_tag[tid] = 0;
  __syncthreads();

  const int ncol = (lane & 15);
  const int koff = (lane < 16) ? 0 : 16;   // B-fragment K offset per lane half
  const int j    = w * 16 + ncol;          // this thread's hidden column 0..255

  for (int t = 0; t < S_; ++t) {
    // ---- stage x_t (f32 -> f16) into LDS --------------------------------
    for (int i = tid; i < 16 * D_; i += 512) {
      int m = i >> 7, k = i & 127;
      sh_x[i] = (_Float16)inputs[(size_t)(b0 + m) * (S_ * D_) + (size_t)t * D_ + k];
    }
    if (t + 1 < S_)  // warm L2/WGP$ for next step's activations
      __builtin_prefetch(inputs + (size_t)(b0 + (tid & 15)) * (S_ * D_) + (size_t)(t + 1) * D_, 0, 1);
    __syncthreads();

    // Launder weight pointers: prevents LICM from hoisting all 36 B-fragments
    // out of the t-loop into registers (288 VGPRs -> scratch spills). Rebuilt
    // as address_space(1) pointers so loads stay global_load_b128, not flat.
    unsigned long long wxa  = (unsigned long long)wx;
    unsigned long long whha = (unsigned long long)whh;
    asm volatile("" : "+s"(wxa), "+s"(whha));
    gf16p wxp  = (gf16p)wxa;
    gf16p whhp = (gf16p)whha;

    v8f acc_gi[3];
    v8f acc_gh[3];
    const v8f zacc = {0.f, 0.f, 0.f, 0.f, 0.f, 0.f, 0.f, 0.f};
    #pragma unroll
    for (int g = 0; g < 3; ++g) { acc_gi[g] = zacc; acc_gh[g] = zacc; }

    // ---- gi = x_t @ Wx^T  (K = 128, 4 WMMA k-steps) ---------------------
    #pragma unroll
    for (int k0 = 0; k0 < D_; k0 += 32) {
      v16h a = load_a_frag(sh_x, D_, k0, lane);
      #pragma unroll
      for (int g = 0; g < 3; ++g) {
        int n = g * H_ + j;                       // gate row 0..767
        v16h bf = *(gv16p)(wxp + n * D_ + k0 + koff);
        acc_gi[g] = __builtin_amdgcn_wmma_f32_16x16x32_f16(
            false, a, false, bf, (short)0, acc_gi[g], false, false);
      }
    }
    // ---- gh = h @ Whh^T  (K = 256, 8 WMMA k-steps) ----------------------
    #pragma unroll
    for (int k0 = 0; k0 < H_; k0 += 32) {
      v16h a = load_a_frag(sh_h, H_, k0, lane);
      #pragma unroll
      for (int g = 0; g < 3; ++g) {
        int n = g * H_ + j;
        v16h bf = *(gv16p)(whhp + n * H_ + k0 + koff);
        acc_gh[g] = __builtin_amdgcn_wmma_f32_16x16x32_f16(
            false, a, false, bf, (short)0, acc_gh[g], false, false);
      }
    }
    __syncthreads();   // all waves done reading sh_h before it is overwritten

    // ---- GRU gate math, fully in accumulator registers ------------------
    // C/D layout: lane gives N=ncol, M = v + (lane<16 ? 0 : 8)
    #pragma unroll
    for (int v = 0; v < 8; ++v) {
      int m  = v + ((lane < 16) ? 0 : 8);
      int tg = sh_tag[m];                        // one-hot column gather
      float gi_r = acc_gi[0][v] + w_ih[(size_t)j * IN_DIM_ + D_ + tg]            + b_ih[j];
      float gh_r = acc_gh[0][v] + b_hh[j];
      float gi_z = acc_gi[1][v] + w_ih[(size_t)(H_ + j) * IN_DIM_ + D_ + tg]     + b_ih[H_ + j];
      float gh_z = acc_gh[1][v] + b_hh[H_ + j];
      float gi_n = acc_gi[2][v] + w_ih[(size_t)(2 * H_ + j) * IN_DIM_ + D_ + tg] + b_ih[2 * H_ + j];
      float gh_n = acc_gh[2][v] + b_hh[2 * H_ + j];
      float r  = 1.0f / (1.0f + __expf(-(gi_r + gh_r)));
      float z  = 1.0f / (1.0f + __expf(-(gi_z + gh_z)));
      float nn = tanhf(gi_n + r * gh_n);
      float hold = sh_hf[m * H_ + j];
      float hnew = (1.0f - z) * nn + z * hold;
      sh_hf[m * H_ + j] = hnew;
      sh_h[m * H_ + j]  = (_Float16)hnew;
    }
    __syncthreads();

    // ---- classifier: logits = h_new @ w_cf^T + b_cf (2 classes) ---------
    if (tid < 32) sh_logits[tid] = b_cf[tid & 1];  // sh_logits[m*2+c]
    __syncthreads();
    {
      int m = tid >> 5, ch = tid & 31;             // 32 chunks of 8 per row
      float p0 = 0.0f, p1 = 0.0f;
      #pragma unroll
      for (int jj = 0; jj < 8; ++jj) {
        int jc = ch * 8 + jj;
        float hv = sh_hf[m * H_ + jc];
        p0 += hv * w_cf[jc];
        p1 += hv * w_cf[H_ + jc];
      }
      atomicAdd(&sh_logits[m * 2 + 0], p0);   // ds_add_f32
      atomicAdd(&sh_logits[m * 2 + 1], p1);
    }
    __syncthreads();
    if (tid < 16) {
      int m = tid;
      float l0 = sh_logits[m * 2 + 0], l1 = sh_logits[m * 2 + 1];
      int act = (l1 > l0) ? 1 : 0;              // argmax, first-max on ties
      float mx = fmaxf(l0, l1);
      float e0 = __expf(l0 - mx), e1 = __expf(l1 - mx);
      float inv = 1.0f / (e0 + e1);
      float p1v = e1 * inv;
      float pi  = act ? p1v : e0 * inv;         // prob of chosen class
      size_t gi = (size_t)(b0 + m) * S_ + t;
      out[gi]                              = (float)act;  // outs
      out[(size_t)B_ * S_ + gi]            = pi;          // pis
      out[(size_t)2 * B_ * S_ + gi]        = p1v;         // probs
      sh_tag[m] += act;
    }
    __syncthreads();
  }
}

// ---------------------------------------------------------------------------
extern "C" void kernel_launch(void* const* d_in, const int* in_sizes, int n_in,
                              void* d_out, int out_size, void* d_ws, size_t ws_size,
                              hipStream_t stream) {
  const float* inputs = (const float*)d_in[0];   // (B,S,D)
  const float* w_ih   = (const float*)d_in[1];   // (768,641)
  const float* w_hh   = (const float*)d_in[2];   // (768,256)
  const float* b_ih   = (const float*)d_in[3];   // (768)
  const float* b_hh   = (const float*)d_in[4];   // (768)
  const float* w_cf   = (const float*)d_in[5];   // (2,256)
  const float* b_cf   = (const float*)d_in[6];   // (2)
  float* out = (float*)d_out;

  _Float16* wx  = (_Float16*)d_ws;               // 768*128 f16
  _Float16* whh = wx + G3_ * D_;                 // 768*256 f16 (total ~576 KB)

  const int prep_elems = G3_ * D_ + G3_ * H_;    // 294912
  prep_weights_kernel<<<(prep_elems + 255) / 256, 256, 0, stream>>>(w_ih, w_hh, wx, whh);

  policy_gru_kernel<<<B_ / 16, 512, 0, stream>>>(inputs, w_ih, b_ih, b_hh,
                                                 w_cf, b_cf, wx, whh, out);
}
